// RangeViT_KPConv_39092792328377
// MI455X (gfx1250) — compile-verified, matrix operand order checked by reading, not run on hardware
//
#include <hip/hip_runtime.h>
#include <math.h>

// ---------------- types ----------------
typedef __attribute__((ext_vector_type(16))) __bf16 v16bf;
typedef __attribute__((ext_vector_type(8)))  float  v8f;
typedef __attribute__((ext_vector_type(4)))  __bf16 v4bf;
typedef int vi4 __attribute__((vector_size(16)));   // native int vector for async-LDS builtin

union BU { v16bf v; uint4 q[2]; };

#define NPTS 262144
#define DCH  128
#define NCLS 20
#define MPROT 10
#define KM   200           // NCLS*MPROT
#define KMPAD 208          // padded cols (13 * 16)
#define BSTRIDE 136        // padded LDS row stride (bf16 elems): 272B, 16B aligned, conflict-free
#define INV_EPS 20.0f      // 1/0.05

// CDNA5 async global->LDS copy support (ASYNCcnt-tracked, bypasses VGPRs)
#if defined(__gfx1250__) && __has_builtin(__builtin_amdgcn_global_load_async_to_lds_b128)
#define HAVE_ASYNC_LDS 1
#else
#define HAVE_ASYNC_LDS 0
#endif

__device__ __forceinline__ void wait_asynccnt0() {
#if defined(__gfx1250__)
#if __has_builtin(__builtin_amdgcn_s_wait_asynccnt)
    __builtin_amdgcn_s_wait_asynccnt(0);
#else
    asm volatile("s_wait_asynccnt 0" ::: "memory");
#endif
#endif
}

// ---------------- wave helpers (wave32) ----------------
__device__ __forceinline__ float wsum(float v) {
    #pragma unroll
    for (int o = 16; o > 0; o >>= 1) v += __shfl_xor(v, o, 32);
    return v;
}
// first-occurrence argmax across wave (ties -> lowest index), matches jnp.argmax
__device__ __forceinline__ void wargmax(float v, int i, float& bv, int& bi) {
    bv = v; bi = i;
    #pragma unroll
    for (int o = 16; o > 0; o >>= 1) {
        float ov = __shfl_xor(bv, o, 32);
        int   oi = __shfl_xor(bi, o, 32);
        if (ov > bv || (ov == bv && oi < bi)) { bv = ov; bi = oi; }
    }
}

// ---------------- init scratch state ----------------
__global__ void k_init(float* s_class, int* bcount, float* S, float* r,
                       float* f, float* ncount) {
    int t = threadIdx.x;
    if (t < NCLS) { s_class[t] = 0.f; bcount[t] = 0; }
    if (t < KM)   { S[t] = 0.f; r[t] = 1.f; ncount[t] = 0.f; }
    for (int i = t; i < KM * DCH; i += 256) f[i] = 0.f;
}

// ---------------- kernel 1: LayerNorm + L2 normalize, split to bf16 hi/lo ----------------
__global__ __launch_bounds__(256) void k_norm(const float* __restrict__ feat,
                                              const float* __restrict__ g,
                                              const float* __restrict__ b,
                                              __bf16* __restrict__ cHi,
                                              __bf16* __restrict__ cLo) {
    int tid = threadIdx.x, lane = tid & 31;
    int n = blockIdx.x * 8 + (tid >> 5);
    const float4 x = *(const float4*)(feat + (size_t)n * DCH + lane * 4);
    float mu = wsum(x.x + x.y + x.z + x.w) * (1.f / 128.f);
    float d0 = x.x - mu, d1 = x.y - mu, d2 = x.z - mu, d3 = x.w - mu;
    float var = wsum(d0 * d0 + d1 * d1 + d2 * d2 + d3 * d3) * (1.f / 128.f);
    float inv = 1.f / sqrtf(var + 1e-5f);
    const float4 gv = *(const float4*)(g + lane * 4);
    const float4 bv = *(const float4*)(b + lane * 4);
    float y0 = d0 * inv * gv.x + bv.x, y1 = d1 * inv * gv.y + bv.y;
    float y2 = d2 * inv * gv.z + bv.z, y3 = d3 * inv * gv.w + bv.w;
    float n2 = wsum(y0 * y0 + y1 * y1 + y2 * y2 + y3 * y3);
    float i2 = 1.f / fmaxf(sqrtf(n2), 1e-12f);
    float c0 = y0 * i2, c1 = y1 * i2, c2 = y2 * i2, c3 = y3 * i2;
    __bf16 h0 = (__bf16)c0, h1 = (__bf16)c1, h2 = (__bf16)c2, h3 = (__bf16)c3;
    v4bf hv = {h0, h1, h2, h3};
    v4bf lv = {(__bf16)(c0 - (float)h0), (__bf16)(c1 - (float)h1),
               (__bf16)(c2 - (float)h2), (__bf16)(c3 - (float)h3)};
    *(v4bf*)(cHi + (size_t)n * DCH + lane * 4) = hv;
    *(v4bf*)(cLo + (size_t)n * DCH + lane * 4) = lv;
}

// ---------------- kernel 1b: normalize prototypes, write f32 + padded bf16 hi/lo ----------------
__global__ void k_protos(const float* __restrict__ prot, float* __restrict__ protoN,
                         __bf16* __restrict__ pH, __bf16* __restrict__ pL) {
    int tid = threadIdx.x, lane = tid & 31;
    int w = blockIdx.x * 8 + (tid >> 5);       // 0..207 (26 blocks)
    float4 x = {0.f, 0.f, 0.f, 0.f};
    if (w < KM) x = *(const float4*)(prot + (size_t)w * DCH + lane * 4);
    float n2 = wsum(x.x * x.x + x.y * x.y + x.z * x.z + x.w * x.w);
    float inv = 1.f / fmaxf(sqrtf(n2), 1e-12f);
    float y0 = x.x * inv, y1 = x.y * inv, y2 = x.z * inv, y3 = x.w * inv;
    if (w < KM) {
        float4 y = {y0, y1, y2, y3};
        *(float4*)(protoN + (size_t)w * DCH + lane * 4) = y;
    }
    __bf16 h0 = (__bf16)y0, h1 = (__bf16)y1, h2 = (__bf16)y2, h3 = (__bf16)y3;
    v4bf hv = {h0, h1, h2, h3};
    v4bf lv = {(__bf16)(y0 - (float)h0), (__bf16)(y1 - (float)h1),
               (__bf16)(y2 - (float)h2), (__bf16)(y3 - (float)h3)};
    size_t base = (size_t)w * BSTRIDE + lane * 4;
    *(v4bf*)(pH + base) = hv;
    *(v4bf*)(pL + base) = lv;
    if (lane == 0) {                           // zero the column pad 128..135
        for (int j = 0; j < BSTRIDE - DCH; j++) {
            pH[(size_t)w * BSTRIDE + DCH + j] = (__bf16)0.f;
            pL[(size_t)w * BSTRIDE + DCH + j] = (__bf16)0.f;
        }
    }
}

// ---------------- kernel 2: WMMA GEMM, bf16x3 error-compensated ----------------
// logits[N,200] = _c[N,128] @ protoN^T[128,200]
__global__ __launch_bounds__(256) void k_gemm(const __bf16* __restrict__ cHi,
                                              const __bf16* __restrict__ cLo,
                                              const __bf16* __restrict__ pH,
                                              const __bf16* __restrict__ pL,
                                              float* __restrict__ logits) {
    __shared__ __bf16 sH[KMPAD * BSTRIDE];
    __shared__ __bf16 sL[KMPAD * BSTRIDE];
    int tid = threadIdx.x;
    {   // stage B (hi and lo) into LDS
        const int nchunk = KMPAD * BSTRIDE * 2 / 16;   // 3536 x 16B per buffer
#if HAVE_ASYNC_LDS
        typedef __attribute__((address_space(1))) vi4 g_vi4;
        typedef __attribute__((address_space(3))) vi4 l_vi4;
        g_vi4* gH = (g_vi4*)pH;
        g_vi4* gL = (g_vi4*)pL;
        l_vi4* lH = (l_vi4*)sH;
        l_vi4* lL = (l_vi4*)sL;
        for (int i = tid; i < nchunk; i += 256) {
            __builtin_amdgcn_global_load_async_to_lds_b128(gH + i, lH + i, 0, 0);
            __builtin_amdgcn_global_load_async_to_lds_b128(gL + i, lL + i, 0, 0);
        }
        wait_asynccnt0();
#else
        const uint4* gH = (const uint4*)pH;
        const uint4* gL = (const uint4*)pL;
        uint4* lH = (uint4*)sH;
        uint4* lL = (uint4*)sL;
        for (int i = tid; i < nchunk; i += 256) { lH[i] = gH[i]; lL[i] = gL[i]; }
#endif
    }
    __syncthreads();

    int lane = tid & 31, wv = tid >> 5, hi = lane >> 4, lm = lane & 15;
    int rowW = blockIdx.x * 128 + wv * 16;
    const __bf16* aHrow = cHi + (size_t)(rowW + lm) * DCH;
    const __bf16* aLrow = cLo + (size_t)(rowW + lm) * DCH;

    v8f acc[13] = {};
    #pragma unroll
    for (int k0 = 0; k0 < DCH; k0 += 32) {
        BU aH, aL;
        aH.q[0] = *(const uint4*)(aHrow + k0 + 8 * hi);
        aH.q[1] = *(const uint4*)(aHrow + k0 + 16 + 8 * hi);
        aL.q[0] = *(const uint4*)(aLrow + k0 + 8 * hi);
        aL.q[1] = *(const uint4*)(aLrow + k0 + 16 + 8 * hi);
        #pragma unroll
        for (int t = 0; t < 13; t++) {
            int col = t * 16 + lm;
            const __bf16* bh = sH + col * BSTRIDE + k0 + 8 * hi;
            const __bf16* bl = sL + col * BSTRIDE + k0 + 8 * hi;
            BU bH, bL;
            bH.q[0] = *(const uint4*)bh; bH.q[1] = *(const uint4*)(bh + 16);
            bL.q[0] = *(const uint4*)bl; bL.q[1] = *(const uint4*)(bl + 16);
            acc[t] = __builtin_amdgcn_wmma_f32_16x16x32_bf16(false, aH.v, false, bH.v, (short)0, acc[t], false, false);
            acc[t] = __builtin_amdgcn_wmma_f32_16x16x32_bf16(false, aH.v, false, bL.v, (short)0, acc[t], false, false);
            acc[t] = __builtin_amdgcn_wmma_f32_16x16x32_bf16(false, aL.v, false, bH.v, (short)0, acc[t], false, false);
        }
    }
    #pragma unroll
    for (int t = 0; t < 13; t++) {
        int col = t * 16 + lm;
        if (col < KM) {
            #pragma unroll
            for (int i = 0; i < 8; i++)
                logits[(size_t)(rowW + 8 * hi + i) * KM + col] = acc[t][i];
        }
    }
}

// ---------------- kernel 3: per-point epilogue ----------------
// max over m -> LayerNorm(20) -> out_seg; pred/correct; e = exp(logits_gt/eps); class sums
__global__ void k_epilogue(const float* __restrict__ logits, const int* __restrict__ gt,
                           const float* __restrict__ mg, const float* __restrict__ mb,
                           float* __restrict__ out_seg, float* __restrict__ e,
                           float* __restrict__ corr, float* __restrict__ s_class,
                           int* __restrict__ bcount) {
    __shared__ float sS[NCLS];
    __shared__ int   sB[NCLS];
    int tid = threadIdx.x;
    if (tid < NCLS) { sS[tid] = 0.f; sB[tid] = 0; }
    __syncthreads();

    int lane = tid & 31;
    int n = blockIdx.x * 8 + (tid >> 5);
    const float* Lrow = logits + (size_t)n * KM;

    float mval = -INFINITY;
    if (lane < NCLS) {
        const float* p = Lrow + lane * MPROT;
        float mv = p[0];
        #pragma unroll
        for (int j = 1; j < MPROT; j++) mv = fmaxf(mv, p[j]);
        mval = mv;
    }
    float xm = (lane < NCLS) ? mval : 0.f;
    float mu = wsum(xm) * (1.f / 20.f);
    float dd = (lane < NCLS) ? (mval - mu) * (mval - mu) : 0.f;
    float var = wsum(dd) * (1.f / 20.f);
    float inv = 1.f / sqrtf(var + 1e-5f);
    float seg = -INFINITY;
    if (lane < NCLS) {
        seg = (mval - mu) * inv * mg[lane] + mb[lane];
        out_seg[(size_t)n * NCLS + lane] = seg;
    }
    float bv; int bi;
    wargmax(seg, lane, bv, bi);
    int gtn = gt[n];
    float cf = (bi == gtn) ? 1.f : 0.f;
    if (lane == 0) corr[n] = cf;

    float ev = 0.f;
    if (lane < MPROT) {
        ev = expf(Lrow[gtn * MPROT + lane] * INV_EPS);
        e[(size_t)n * MPROT + lane] = ev;
    }
    float es = wsum(ev);
    if (lane == 0) { atomicAdd(&sS[gtn], es); atomicAdd(&sB[gtn], 1); }
    __syncthreads();
    if (tid < NCLS) { atomicAdd(&s_class[tid], sS[tid]); atomicAdd(&bcount[tid], sB[tid]); }
}

// ---------------- sinkhorn kernels (factorized L = E * r[k,m] * c[n]) ----------------
__global__ void k_sinkA0(const float* __restrict__ e, const int* __restrict__ gt,
                         const float* __restrict__ s_class, float* __restrict__ cvec,
                         float* __restrict__ S) {
    __shared__ float sS[KM];
    for (int i = threadIdx.x; i < KM; i += 256) sS[i] = 0.f;
    __syncthreads();
    int n = blockIdx.x * 256 + threadIdx.x;
    int k = gt[n];
    float s = s_class[k];
    float c = 1.f / ((s > 0.f) ? s : 1.f);
    cvec[n] = c;
    int base = k * MPROT;
    #pragma unroll
    for (int m = 0; m < MPROT; m++) atomicAdd(&sS[base + m], e[(size_t)n * MPROT + m] * c);
    __syncthreads();
    for (int i = threadIdx.x; i < KM; i += 256) atomicAdd(&S[i], sS[i]);
}

__global__ void k_sinkB(float* __restrict__ r, float* __restrict__ S) {
    int i = threadIdx.x;
    if (i < KM) {
        float rv = r[i];
        float rs = rv * S[i];
        r[i] = rv / (((rs > 0.f) ? rs : 1.f) * (float)MPROT);
        S[i] = 0.f;
    }
}

__global__ void k_sinkCA(const float* __restrict__ e, const int* __restrict__ gt,
                         const int* __restrict__ bcount, const float* __restrict__ r,
                         float* __restrict__ cvec, float* __restrict__ S) {
    __shared__ float sS[KM];
    for (int i = threadIdx.x; i < KM; i += 256) sS[i] = 0.f;
    __syncthreads();
    int n = blockIdx.x * 256 + threadIdx.x;
    int k = gt[n];
    float ev[MPROT];
    float t = 0.f;
    #pragma unroll
    for (int m = 0; m < MPROT; m++) {
        ev[m] = e[(size_t)n * MPROT + m];
        t += ev[m] * r[k * MPROT + m];
    }
    float c = cvec[n];
    float cs = c * t;
    int bc = bcount[k];
    float Bs = (bc > 0) ? (float)bc : 1.f;
    float cn = (cs > 0.f) ? 1.f / (t * Bs) : c / Bs;
    cvec[n] = cn;
    #pragma unroll
    for (int m = 0; m < MPROT; m++) atomicAdd(&sS[k * MPROT + m], ev[m] * cn);
    __syncthreads();
    for (int i = threadIdx.x; i < KM; i += 256) atomicAdd(&S[i], sS[i]);
}

// ---------------- kernel D: hard assignment + scatter into f / ncount ----------------
__global__ void k_assign(const float* __restrict__ e, const float* __restrict__ r,
                         const int* __restrict__ gt, const float* __restrict__ corr,
                         const __bf16* __restrict__ cHi, const __bf16* __restrict__ cLo,
                         float* __restrict__ ptarget, float* __restrict__ f,
                         float* __restrict__ ncount) {
    int tid = threadIdx.x, lane = tid & 31;
    int n = blockIdx.x * 8 + (tid >> 5);
    int k = gt[n];
    float v = -INFINITY;
    if (lane < MPROT) v = e[(size_t)n * MPROT + lane] * r[k * MPROT + lane];
    float bv; int bi;
    wargmax(v, lane, bv, bi);
    if (lane == 0) ptarget[n] = (float)(bi + MPROT * k);
    if (corr[n] != 0.f) {
        int p = k * MPROT + bi;
        v4bf h = *(const v4bf*)(cHi + (size_t)n * DCH + lane * 4);
        v4bf l = *(const v4bf*)(cLo + (size_t)n * DCH + lane * 4);
        float* fp = f + (size_t)p * DCH + lane * 4;
        atomicAdd(fp + 0, (float)h.x + (float)l.x);
        atomicAdd(fp + 1, (float)h.y + (float)l.y);
        atomicAdd(fp + 2, (float)h.z + (float)l.z);
        atomicAdd(fp + 3, (float)h.w + (float)l.w);
        if (lane == 0) atomicAdd(&ncount[p], 1.f);
    }
}

// ---------------- kernel E: momentum update + renormalize prototypes ----------------
__global__ void k_update(const float* __restrict__ f, const float* __restrict__ ncount,
                         const float* __restrict__ protoN, float* __restrict__ newp) {
    int tid = threadIdx.x, lane = tid & 31;
    int w = blockIdx.x * 8 + (tid >> 5);   // 0..199 (25 blocks)
    const float4 fv = *(const float4*)(f + (size_t)w * DCH + lane * 4);
    float n2 = wsum(fv.x * fv.x + fv.y * fv.y + fv.z * fv.z + fv.w * fv.w);
    float inv = 1.f / fmaxf(sqrtf(n2), 1e-12f);
    const float4 pn = *(const float4*)(protoN + (size_t)w * DCH + lane * 4);
    bool has = ncount[w] > 0.f;
    const float G = 0.99f, OMG = 1.0f - 0.99f;
    float x0 = has ? G * pn.x + OMG * fv.x * inv : pn.x;
    float x1 = has ? G * pn.y + OMG * fv.y * inv : pn.y;
    float x2 = has ? G * pn.z + OMG * fv.z * inv : pn.z;
    float x3 = has ? G * pn.w + OMG * fv.w * inv : pn.w;
    float n2b = wsum(x0 * x0 + x1 * x1 + x2 * x2 + x3 * x3);
    float ib = 1.f / fmaxf(sqrtf(n2b), 1e-12f);
    float4 o = {x0 * ib, x1 * ib, x2 * ib, x3 * ib};
    *(float4*)(newp + (size_t)w * DCH + lane * 4) = o;
}

// ---------------- launch ----------------
extern "C" void kernel_launch(void* const* d_in, const int* in_sizes, int n_in,
                              void* d_out, int out_size, void* d_ws, size_t ws_size,
                              hipStream_t stream) {
    const int N = NPTS;
    const float* feat = (const float*)d_in[0];
    const int*   gt   = (const int*)d_in[1];
    const float* prot = (const float*)d_in[2];
    const float* fg   = (const float*)d_in[3];
    const float* fb   = (const float*)d_in[4];
    const float* mg   = (const float*)d_in[5];
    const float* mb   = (const float*)d_in[6];

    float* out      = (float*)d_out;
    float* out_seg  = out;                                   // [N,20]
    float* logits   = out + (size_t)N * NCLS;                // [N,200]
    float* ptarget  = logits + (size_t)N * KM;               // [N]
    float* newp     = ptarget + N;                           // [200,128]

    char* w = (char*)d_ws;
    size_t off = 0;
    auto alloc = [&](size_t bytes) -> char* {
        char* p = w + off;
        off += (bytes + 255) & ~(size_t)255;
        return p;
    };
    __bf16* cHi     = (__bf16*)alloc((size_t)N * DCH * 2);
    __bf16* cLo     = (__bf16*)alloc((size_t)N * DCH * 2);
    float*  e       = (float*) alloc((size_t)N * MPROT * 4);
    float*  cvec    = (float*) alloc((size_t)N * 4);
    float*  corr    = (float*) alloc((size_t)N * 4);
    float*  protoN  = (float*) alloc((size_t)KM * DCH * 4);
    __bf16* pH      = (__bf16*)alloc((size_t)KMPAD * BSTRIDE * 2);
    __bf16* pL      = (__bf16*)alloc((size_t)KMPAD * BSTRIDE * 2);
    float*  s_class = (float*) alloc(NCLS * 4);
    int*    bcount  = (int*)   alloc(NCLS * 4);
    float*  S       = (float*) alloc(KM * 4);
    float*  r       = (float*) alloc(KM * 4);
    float*  fbuf    = (float*) alloc((size_t)KM * DCH * 4);
    float*  ncount  = (float*) alloc(KM * 4);
    (void)ws_size; (void)in_sizes; (void)n_in; (void)out_size;

    k_init    <<<1, 256, 0, stream>>>(s_class, bcount, S, r, fbuf, ncount);
    k_norm    <<<N / 8, 256, 0, stream>>>(feat, fg, fb, cHi, cLo);
    k_protos  <<<26, 256, 0, stream>>>(prot, protoN, pH, pL);
    k_gemm    <<<N / 128, 256, 0, stream>>>(cHi, cLo, pH, pL, logits);
    k_epilogue<<<N / 8, 256, 0, stream>>>(logits, gt, mg, mb, out_seg, e, corr, s_class, bcount);
    // Sinkhorn (3 iterations): A0 computes S with c0; B updates r (+zero S); CA updates c and next S
    k_sinkA0  <<<N / 256, 256, 0, stream>>>(e, gt, s_class, cvec, S);
    k_sinkB   <<<1, 256, 0, stream>>>(r, S);
    k_sinkCA  <<<N / 256, 256, 0, stream>>>(e, gt, bcount, r, cvec, S);
    k_sinkB   <<<1, 256, 0, stream>>>(r, S);
    k_sinkCA  <<<N / 256, 256, 0, stream>>>(e, gt, bcount, r, cvec, S);
    k_sinkB   <<<1, 256, 0, stream>>>(r, S);
    k_assign  <<<N / 8, 256, 0, stream>>>(e, r, gt, corr, cHi, cLo, ptarget, fbuf, ncount);
    k_update  <<<25, 256, 0, stream>>>(fbuf, ncount, protoN, newp);
}
